// Model_25907242729615
// MI455X (gfx1250) — compile-verified, matrix operand order checked by reading, not run on hardware
//
#include <hip/hip_runtime.h>
#include <cstdint>
#include <cstddef>

// ---------------------------------------------------------------------------
// GVP-GNN forward for MI455X (gfx1250, wave32, WMMA).
// All dense contractions run through v_wmma_f32_16x16x32_f16.
// convs[0] and the final vector channel are skipped (dead in the reference).
// ---------------------------------------------------------------------------

typedef __attribute__((ext_vector_type(16))) _Float16 v16h;
typedef __attribute__((ext_vector_type(8)))  float    v8f;

#define EPSN 1e-8f

__device__ __forceinline__ void fadd(float* p, float v) {
  __hip_atomic_fetch_add(p, v, __ATOMIC_RELAXED, __HIP_MEMORY_SCOPE_AGENT);
}

// ---------------------------------------------------------------------------
// Generic GEMM: C[M,N] = A[M,K] * B[N,K]^T (+bias, +relu), f16 in, f32 acc.
// Block: 256 threads = 8 waves; block tile 128x128; wave tile 32x64
// => 8 v_wmma_f32_16x16x32_f16 per wave per K-step against 12 ds_load_b128.
// LDS rows padded to 48 halves (96B) so fragment b128 loads stay 16B aligned.
// Fragment layouts follow CDNA5 ISA 7.12.2:
//   A (16x32 f16): lane L -> M=L%16; halves = K {base..base+7, 16+base..}, base=(L<16)?0:8
//   B (32x16 f16): lane L -> N=L%16; same K striping
//   C (16x16 f32): lane L -> N=L%16; elem r -> M = r + ((L<16)?0:8)
// ---------------------------------------------------------------------------
__global__ __launch_bounds__(256) void k_gemm(
    const _Float16* __restrict__ A, int lda, int M,
    const _Float16* __restrict__ B, int ldb, int N, int K,
    const float* __restrict__ bias, int act,
    _Float16* __restrict__ out16, int ldc16,
    float* __restrict__ out32, int ldc32)
{
  __shared__ _Float16 As[128 * 48];
  __shared__ _Float16 Bs[128 * 48];
  const int gm = blockIdx.x * 128, gn = blockIdx.y * 128;
  const int t = threadIdx.x;
  const int w = t >> 5, lane = t & 31;
  const int wm = (w >> 1) << 5;          // 0,32,64,96
  const int wn = (w & 1) << 6;           // 0,64
  const int r = lane & 15;
  const int hsel = (lane < 16) ? 0 : 8;

  const v8f vzero = {0.f, 0.f, 0.f, 0.f, 0.f, 0.f, 0.f, 0.f};
  v8f acc[2][4];
#pragma unroll
  for (int mt = 0; mt < 2; mt++)
#pragma unroll
    for (int nt = 0; nt < 4; nt++) acc[mt][nt] = vzero;

  for (int k0 = 0; k0 < K; k0 += 32) {
    __syncthreads();
    // cooperative staging: each thread moves 2 A-chunks + 2 B-chunks (16B each)
#pragma unroll
    for (int q = 0; q < 2; q++) {
      const int cid = t + q * 256;            // 0..511 over 128 rows x 4 chunks
      const int row = cid >> 2, ch = (cid & 3) << 3;
      uint4 av4 = make_uint4(0u, 0u, 0u, 0u);
      const int arow = gm + row;
      if (arow < M) av4 = *(const uint4*)(A + (size_t)arow * lda + k0 + ch);
      *(uint4*)(As + row * 48 + ch) = av4;
      uint4 bv4 = make_uint4(0u, 0u, 0u, 0u);
      const int brow = gn + row;
      if (brow < N) bv4 = *(const uint4*)(B + (size_t)brow * ldb + k0 + ch);
      *(uint4*)(Bs + row * 48 + ch) = bv4;
    }
    __syncthreads();

    v16h a[2], bf[4];
#pragma unroll
    for (int mt = 0; mt < 2; mt++) {
      const _Float16* p = As + (wm + mt * 16 + r) * 48 + hsel;
      ((uint4*)&a[mt])[0] = *(const uint4*)p;
      ((uint4*)&a[mt])[1] = *(const uint4*)(p + 16);
    }
#pragma unroll
    for (int nt = 0; nt < 4; nt++) {
      const _Float16* p = Bs + (wn + nt * 16 + r) * 48 + hsel;
      ((uint4*)&bf[nt])[0] = *(const uint4*)p;
      ((uint4*)&bf[nt])[1] = *(const uint4*)(p + 16);
    }
#pragma unroll
    for (int mt = 0; mt < 2; mt++)
#pragma unroll
      for (int nt = 0; nt < 4; nt++)
        acc[mt][nt] = __builtin_amdgcn_wmma_f32_16x16x32_f16(
            false, a[mt], false, bf[nt], (short)0, acc[mt][nt], false, false);
  }

#pragma unroll
  for (int mt = 0; mt < 2; mt++) {
    const int mb = gm + wm + mt * 16 + ((lane < 16) ? 0 : 8);
#pragma unroll
    for (int nt = 0; nt < 4; nt++) {
      const int n = gn + wn + nt * 16 + r;
      if (n >= N) continue;
      const float bsv = bias ? bias[n] : 0.f;
#pragma unroll
      for (int e = 0; e < 8; e++) {
        const int m = mb + e;
        if (m >= M) continue;
        float vv = acc[mt][nt][e] + bsv;
        if (act == 1) vv = fmaxf(vv, 0.f);
        if (out16) out16[(size_t)m * ldc16 + n] = (_Float16)vv;
        if (out32) out32[(size_t)m * ldc32 + n] = vv;
      }
    }
  }
}

// fp32 weight [N,K] -> f16 [N,Kpad] zero padded
__global__ void k_w16(const float* __restrict__ src, _Float16* __restrict__ dst,
                      int K, int Kpad)
{
  const int row = blockIdx.y;
  const int col = blockIdx.x * 256 + threadIdx.x;
  if (col >= Kpad) return;
  dst[(size_t)row * Kpad + col] = (col < K) ? (_Float16)src[(size_t)row * K + col]
                                            : (_Float16)0.f;
}

// Encoder GVP front half (vi=1): vh[n,h,c] = wh[h]*v[n,0,c]; scat = [s | vn | pad]
__global__ void k_enc(const float* __restrict__ s, const float* __restrict__ v,
                      const float* __restrict__ wh,
                      _Float16* __restrict__ vh, size_t pst,
                      _Float16* __restrict__ scat, int n)
{
  const int i = blockIdx.x * 256 + threadIdx.x;
  const int node = i >> 7, h = i & 127;
  if (node >= n) return;
  const float r0 = v[node * 3 + 0], r1 = v[node * 3 + 1], r2 = v[node * 3 + 2];
  const float w = wh[h];
  const size_t o = (size_t)node * 128 + h;
  vh[o]           = (_Float16)(w * r0);
  vh[pst + o]     = (_Float16)(w * r1);
  vh[2 * pst + o] = (_Float16)(w * r2);
  const float vn = sqrtf(fmaxf(w * w * (r0 * r0 + r1 * r1 + r2 * r2), EPSN));
  _Float16* row = scat + (size_t)node * 160;
  row[1 + h] = (_Float16)vn;
  if (h == 0) row[0] = (_Float16)s[node];
  if (h < 31) row[129 + h] = (_Float16)0.f;   // zero the K-pad (129 -> 160)
}

// vn[m,h] = sqrt(max(sum_c vh^2, eps)) -> dst[m*ldd+off+h]. grid=(Hd/128, M)
__global__ void k_vnorm(const _Float16* __restrict__ vh, size_t pst, int Hd,
                        _Float16* __restrict__ dst, int ldd, int off)
{
  const int m = blockIdx.y;
  const int h = blockIdx.x * 128 + threadIdx.x;
  const size_t o = (size_t)m * Hd + h;
  const float x = (float)vh[o], y = (float)vh[pst + o], z = (float)vh[2 * pst + o];
  dst[(size_t)m * ldd + off + h] = (_Float16)sqrtf(fmaxf(x * x + y * y + z * z, EPSN));
}

// vo *= sigmoid(||vo||_3); planes are [M,O] flat, so i indexes within a plane.
__global__ void k_vgate(const _Float16* __restrict__ vo, size_t pst, size_t total,
                        _Float16* __restrict__ o16, float* __restrict__ o32)
{
  const size_t i = (size_t)blockIdx.x * 256 + threadIdx.x;
  if (i >= total) return;
  float x = (float)vo[i], y = (float)vo[pst + i], z = (float)vo[2 * pst + i];
  const float g = 1.f / (1.f + __expf(-sqrtf(fmaxf(x * x + y * y + z * z, EPSN))));
  x *= g; y *= g; z *= g;
  o16[i] = (_Float16)x; o16[pst + i] = (_Float16)y; o16[2 * pst + i] = (_Float16)z;
  if (o32) { o32[i] = x; o32[pst + i] = y; o32[2 * pst + i] = z; }
}

// Build edge message concat: scat0[:, :384] = [s_src|es|s_dst]; mvcat planes same.
__global__ void k_gather(_Float16* __restrict__ scat0, _Float16* __restrict__ mvcat, size_t pst,
                         const _Float16* __restrict__ sN, const _Float16* __restrict__ vN, size_t vpst,
                         const _Float16* __restrict__ es, const _Float16* __restrict__ ev, size_t epst,
                         const int* __restrict__ src, const int* __restrict__ dst,
                         int e0, int c)
{
  const int i = blockIdx.x * 256 + threadIdx.x;
  if (i >= c * 128) return;
  const int il = i >> 7, cc = i & 127;
  const int e = e0 + il;
  const int si = src[e], di = dst[e];
  _Float16* srow = scat0 + (size_t)il * 768;
  srow[cc]       = sN[(size_t)si * 128 + cc];
  srow[128 + cc] = es[(size_t)e * 128 + cc];
  srow[256 + cc] = sN[(size_t)di * 128 + cc];
#pragma unroll
  for (int p = 0; p < 3; p++) {
    _Float16* vrow = mvcat + p * pst + (size_t)il * 384;
    vrow[cc]       = vN[p * vpst + (size_t)si * 128 + cc];
    vrow[128 + cc] = ev[p * epst + (size_t)e * 128 + cc];
    vrow[256 + cc] = vN[p * vpst + (size_t)di * 128 + cc];
  }
}

__global__ void k_count(const int* __restrict__ dst, int E, float* __restrict__ cnt)
{
  const int e = blockIdx.x * 256 + threadIdx.x;
  if (e < E) fadd(&cnt[dst[e]], 1.f);
}

__global__ void k_scatter(const float* __restrict__ ms, const float* __restrict__ mv, size_t pst,
                          const int* __restrict__ dst, int e0, int c,
                          float* __restrict__ as_, float* __restrict__ av, size_t apst)
{
  const int i = blockIdx.x * 256 + threadIdx.x;
  if (i >= c * 128) return;
  const int il = i >> 7, cc = i & 127;
  const int d = dst[e0 + il];
  fadd(&as_[(size_t)d * 128 + cc], ms[(size_t)il * 128 + cc]);
#pragma unroll
  for (int p = 0; p < 3; p++)
    fadd(&av[p * apst + (size_t)d * 128 + cc], mv[p * pst + (size_t)il * 128 + cc]);
}

// residual + layernorm(ln0): one wave32 per node (8 waves / block)
__global__ void k_ln0(const float* __restrict__ sN, const float* __restrict__ as_,
                      const float* __restrict__ vN, const float* __restrict__ av, size_t vpst,
                      const float* __restrict__ cnt,
                      const float* __restrict__ g, const float* __restrict__ b,
                      float* __restrict__ sln32, _Float16* __restrict__ sln16, int ld16,
                      float* __restrict__ vln32, _Float16* __restrict__ vln16, int n)
{
  const int node = blockIdx.x * 8 + (threadIdx.x >> 5);
  const int lane = threadIdx.x & 31;
  if (node >= n) return;
  const float inv = 1.f / fmaxf(cnt[node], 1.f);
  const size_t base = (size_t)node * 128;
  float sv[4]; float mu = 0.f;
#pragma unroll
  for (int j = 0; j < 4; j++) {
    const int col = lane + j * 32;
    sv[j] = sN[base + col] + as_[base + col] * inv;
    mu += sv[j];
  }
#pragma unroll
  for (int o = 16; o > 0; o >>= 1) mu += __shfl_xor(mu, o, 32);
  mu *= (1.f / 128.f);
  float var = 0.f;
#pragma unroll
  for (int j = 0; j < 4; j++) { const float d = sv[j] - mu; var += d * d; }
#pragma unroll
  for (int o = 16; o > 0; o >>= 1) var += __shfl_xor(var, o, 32);
  const float rs = rsqrtf(var * (1.f / 128.f) + 1e-5f);
#pragma unroll
  for (int j = 0; j < 4; j++) {
    const int col = lane + j * 32;
    const float o = (sv[j] - mu) * rs * g[col] + b[col];
    sln32[base + col] = o;
    sln16[(size_t)node * ld16 + col] = (_Float16)o;
  }
  // vector path: v/sqrt(mean_i(clamp(|v_i|^2, eps)))
  float vx[4], vy[4], vz[4]; float ss = 0.f;
#pragma unroll
  for (int j = 0; j < 4; j++) {
    const size_t idx = base + lane + j * 32;
    const float x = vN[idx] + av[idx] * inv;
    const float y = vN[vpst + idx] + av[vpst + idx] * inv;
    const float z = vN[2 * vpst + idx] + av[2 * vpst + idx] * inv;
    ss += fmaxf(x * x + y * y + z * z, EPSN);
    vx[j] = x; vy[j] = y; vz[j] = z;
  }
#pragma unroll
  for (int o = 16; o > 0; o >>= 1) ss += __shfl_xor(ss, o, 32);
  const float sc = rsqrtf(ss * (1.f / 128.f));
#pragma unroll
  for (int j = 0; j < 4; j++) {
    const size_t idx = base + lane + j * 32;
    vln32[idx] = vx[j] * sc;            vln16[idx] = (_Float16)(vx[j] * sc);
    vln32[vpst + idx] = vy[j] * sc;     vln16[vpst + idx] = (_Float16)(vy[j] * sc);
    vln32[2 * vpst + idx] = vz[j] * sc; vln16[2 * vpst + idx] = (_Float16)(vz[j] * sc);
  }
}

// final residual + layernorm(ln1), scalar path only (vector output is dead)
__global__ void k_ln1(const float* __restrict__ sln, const float* __restrict__ ds,
                      const float* __restrict__ g, const float* __restrict__ b,
                      float* __restrict__ xs, int n)
{
  const int node = blockIdx.x * 8 + (threadIdx.x >> 5);
  const int lane = threadIdx.x & 31;
  if (node >= n) return;
  const size_t base = (size_t)node * 128;
  float sv[4]; float mu = 0.f;
#pragma unroll
  for (int j = 0; j < 4; j++) {
    const int col = lane + j * 32;
    sv[j] = sln[base + col] + ds[base + col];
    mu += sv[j];
  }
#pragma unroll
  for (int o = 16; o > 0; o >>= 1) mu += __shfl_xor(mu, o, 32);
  mu *= (1.f / 128.f);
  float var = 0.f;
#pragma unroll
  for (int j = 0; j < 4; j++) { const float d = sv[j] - mu; var += d * d; }
#pragma unroll
  for (int o = 16; o > 0; o >>= 1) var += __shfl_xor(var, o, 32);
  const float rs = rsqrtf(var * (1.f / 128.f) + 1e-5f);
#pragma unroll
  for (int j = 0; j < 4; j++) {
    const int col = lane + j * 32;
    xs[base + col] = (sv[j] - mu) * rs * g[col] + b[col];
  }
}

__global__ void k_pool(const float* __restrict__ xs, const int* __restrict__ batch,
                       float* __restrict__ gsum, float* __restrict__ gcnt, int n)
{
  const int i = blockIdx.x * 256 + threadIdx.x;
  if (i >= n * 128) return;
  const int node = i >> 7, cc = i & 127;
  const int gid = batch[node];
  fadd(&gsum[(size_t)gid * 128 + cc], xs[i]);
  if (cc == 0) fadd(&gcnt[gid], 1.f);
}

__global__ void k_mlp(const float* __restrict__ gsum, const float* __restrict__ gcnt,
                      const float* __restrict__ w0, const float* __restrict__ b0,
                      const float* __restrict__ w1, const float* __restrict__ b1,
                      const float* __restrict__ w2, const float* __restrict__ b2,
                      float* __restrict__ out)
{
  const int g = threadIdx.x;
  if (g >= 64) return;
  const float inv = 1.f / fmaxf(gcnt[g], 1.f);
  float x[128];
  for (int c = 0; c < 128; c++) x[c] = gsum[(size_t)g * 128 + c] * inv;
  float h0[64];
  for (int o = 0; o < 64; o++) {
    float a = b0[o];
    for (int c = 0; c < 128; c++) a += w0[o * 128 + c] * x[c];
    h0[o] = fmaxf(a, 0.f);
  }
  float h1[64];
  for (int o = 0; o < 64; o++) {
    float a = b1[o];
    for (int c = 0; c < 64; c++) a += w1[o * 64 + c] * h0[c];
    h1[o] = fmaxf(a, 0.f);
  }
  float a = b2[0];
  for (int c = 0; c < 64; c++) a += w2[c] * h1[c];
  out[g] = 1.f / (1.f + __expf(-a));
}

// ---------------------------------------------------------------------------
// Host orchestration
// ---------------------------------------------------------------------------
extern "C" void kernel_launch(void* const* d_in, const int* in_sizes, int n_in,
                              void* d_out, int out_size, void* d_ws, size_t ws_size,
                              hipStream_t stream)
{
  (void)in_sizes; (void)n_in; (void)out_size; (void)ws_size;
  constexpr int NN = 10000, NE = 160000, CH = 8192;
  constexpr int CB = 36;   // first leaf of convs[1] (only conv whose output survives)

  // Input leaf map (dict-insertion order flattening):
  // 0 node_s, 1 node_v, 2 edge_s, 3 edge_v,
  // 4..7   node_enc (wh, ws.w, ws.b, wv)
  // 8..11  edge_enc
  // 12..35 convs[0]  (skipped: dead in reference)
  // 36..59 convs[1]: msg0 +0..3, msg1 +4..7, msg2 +8..11, ln0 +12..13,
  //                  ff0 +14..17, ff1 +18..21, ln1 +22..23
  // 60..65 mlp w0,b0,w1,b1,w2,b2;  66 edge_index, 67 batch
  auto P = [&](int i) { return (const float*)d_in[i]; };
  const float* node_s = P(0); const float* node_v = P(1);
  const float* edge_s = P(2); const float* edge_v = P(3);
  const int* ei   = (const int*)d_in[66];
  const int* srcp = ei, *dstp = ei + NE;
  const int* batch = (const int*)d_in[67];

  size_t cur = 0;
  auto alloc = [&](size_t bytes) -> void* {
    cur = (cur + 255) & ~(size_t)255;
    void* p = (char*)d_ws + cur;
    cur += bytes;
    return p;
  };
  auto gemm = [&](const _Float16* A, int lda, int M, const _Float16* B, int ldb,
                  int N_, int K, const float* bias, int act,
                  _Float16* o16, int ldc16, float* o32, int ldc32) {
    dim3 grid((M + 127) / 128, (N_ + 127) / 128);
    k_gemm<<<grid, 256, 0, stream>>>(A, lda, M, B, ldb, N_, K, bias, act,
                                     o16, ldc16, o32, ldc32);
  };
  auto w16 = [&](int idx, int N_, int K_, int Kpad) -> _Float16* {
    _Float16* dst = (_Float16*)alloc((size_t)N_ * Kpad * 2);
    dim3 grid((Kpad + 255) / 256, N_);
    k_w16<<<grid, 256, 0, stream>>>((const float*)d_in[idx], dst, K_, Kpad);
    return dst;
  };

  // ---- weights -> f16 (K padded to mult of 32) ----
  _Float16* wsN = w16(5,  128, 129, 160);
  _Float16* wvN = w16(7,  128, 128, 128);
  _Float16* wsE = w16(9,  128, 129, 160);
  _Float16* wvE = w16(11, 128, 128, 128);
  _Float16* m0wh = w16(CB + 0,  384, 384, 384);
  _Float16* m0ws = w16(CB + 1,  128, 768, 768);
  _Float16* m0wv = w16(CB + 3,  128, 384, 384);
  _Float16* m1wh = w16(CB + 4,  128, 128, 128);
  _Float16* m1ws = w16(CB + 5,  128, 256, 256);
  _Float16* m1wv = w16(CB + 7,  128, 128, 128);
  _Float16* m2wh = w16(CB + 8,  128, 128, 128);
  _Float16* m2ws = w16(CB + 9,  128, 256, 256);
  _Float16* m2wv = w16(CB + 11, 128, 128, 128);
  _Float16* f0wh = w16(CB + 14, 256, 128, 128);
  _Float16* f0ws = w16(CB + 15, 512, 384, 384);
  _Float16* f0wv = w16(CB + 17, 256, 256, 256);
  _Float16* f1wh = w16(CB + 18, 256, 256, 256);
  _Float16* f1ws = w16(CB + 19, 128, 768, 768);
  // f1wv (CB+21) skipped: ff1 vector output is dead.

  // ---- persistent activations ----
  float*     sN32 = (float*)alloc((size_t)NN * 128 * 4);
  _Float16*  sN16 = (_Float16*)alloc((size_t)NN * 128 * 2);
  float*     vN32 = (float*)alloc(3ull * NN * 128 * 4);
  _Float16*  vN16 = (_Float16*)alloc(3ull * NN * 128 * 2);
  _Float16*  es16 = (_Float16*)alloc((size_t)NE * 128 * 2);
  _Float16*  ev16 = (_Float16*)alloc(3ull * NE * 128 * 2);
  float*     asum = (float*)alloc((size_t)NN * 128 * 4);
  float*     avum = (float*)alloc(3ull * NN * 128 * 4);
  float*     cntb = (float*)alloc((size_t)NN * 4);
  float*     sln32 = (float*)alloc((size_t)NN * 128 * 4);
  _Float16*  scatF0 = (_Float16*)alloc((size_t)NN * 384 * 2);  // [s_ln | vn_ff0]
  float*     vln32 = (float*)alloc(3ull * NN * 128 * 4);
  _Float16*  vln16 = (_Float16*)alloc(3ull * NN * 128 * 2);
  float*     ds1  = (float*)alloc((size_t)NN * 128 * 4);
  float*     xs   = (float*)alloc((size_t)NN * 128 * 4);
  float*     gsum = (float*)alloc(64 * 128 * 4);
  float*     gcnt = (float*)alloc(64 * 4);
  const size_t persist_end = cur;

  // zero the accumulators each call (graph-capture-safe memsets)
  hipMemsetAsync(asum, 0, (size_t)NN * 128 * 4, stream);
  hipMemsetAsync(avum, 0, 3ull * NN * 128 * 4, stream);
  hipMemsetAsync(cntb, 0, (size_t)NN * 4, stream);
  hipMemsetAsync(gsum, 0, 64 * 128 * 4, stream);
  hipMemsetAsync(gcnt, 0, 64 * 4, stream);

  // ---- phase 1 scratch: encoders ----
  {
    _Float16* scatN = (_Float16*)alloc((size_t)NN * 160 * 2);
    _Float16* vhN   = (_Float16*)alloc(3ull * NN * 128 * 2);
    _Float16* voN   = (_Float16*)alloc(3ull * NN * 128 * 2);
    k_enc<<<(NN * 128 + 255) / 256, 256, 0, stream>>>(node_s, node_v, P(4), vhN,
                                                      (size_t)NN * 128, scatN, NN);
    gemm(scatN, 160, NN, wsN, 160, 128, 160, P(6), 1, sN16, 128, sN32, 128);
    gemm(vhN, 128, 3 * NN, wvN, 128, 128, 128, nullptr, 0, voN, 128, nullptr, 0);
    k_vgate<<<((size_t)NN * 128 + 255) / 256, 256, 0, stream>>>(
        voN, (size_t)NN * 128, (size_t)NN * 128, vN16, vN32);

    _Float16* scatE = (_Float16*)alloc((size_t)NE * 160 * 2);
    _Float16* vhE   = (_Float16*)alloc(3ull * NE * 128 * 2);
    _Float16* voE   = (_Float16*)alloc(3ull * NE * 128 * 2);
    k_enc<<<(NE * 128 + 255) / 256, 256, 0, stream>>>(edge_s, edge_v, P(8), vhE,
                                                      (size_t)NE * 128, scatE, NE);
    gemm(scatE, 160, NE, wsE, 160, 128, 160, P(10), 1, es16, 128, nullptr, 0);
    gemm(vhE, 128, 3 * NE, wvE, 128, 128, 128, nullptr, 0, voE, 128, nullptr, 0);
    k_vgate<<<((size_t)NE * 128 + 255) / 256, 256, 0, stream>>>(
        voE, (size_t)NE * 128, (size_t)NE * 128, ev16, nullptr);
  }

  k_count<<<(NE + 255) / 256, 256, 0, stream>>>(dstp, NE, cntb);

  // ---- phase 2 scratch (overlays phase 1; stream-ordered so this is safe) ----
  cur = persist_end;
  _Float16* scat0 = (_Float16*)alloc((size_t)CH * 768 * 2);  // [ms | vn0]
  _Float16* mvcat = (_Float16*)alloc(3ull * CH * 384 * 2);
  _Float16* vh0   = (_Float16*)alloc(3ull * CH * 384 * 2);
  _Float16* scat1 = (_Float16*)alloc((size_t)CH * 256 * 2);  // [s1 | vn1]
  _Float16* vt0   = (_Float16*)alloc(3ull * CH * 128 * 2);
  _Float16* v1    = (_Float16*)alloc(3ull * CH * 128 * 2);
  _Float16* vh1   = (_Float16*)alloc(3ull * CH * 128 * 2);
  _Float16* scat2 = (_Float16*)alloc((size_t)CH * 256 * 2);  // [s2 | vn2]
  _Float16* v2    = (_Float16*)alloc(3ull * CH * 128 * 2);
  _Float16* vh2   = (_Float16*)alloc(3ull * CH * 128 * 2);
  float*    msO   = (float*)alloc((size_t)CH * 128 * 4);
  float*    mvO   = (float*)alloc(3ull * CH * 128 * 4);

  for (int e0 = 0; e0 < NE; e0 += CH) {
    const int c = (NE - e0 < CH) ? (NE - e0) : CH;
    const size_t p384 = (size_t)c * 384, p128 = (size_t)c * 128;
    k_gather<<<(c * 128 + 255) / 256, 256, 0, stream>>>(
        scat0, mvcat, p384, sN16, vN16, (size_t)NN * 128, es16, ev16,
        (size_t)NE * 128, srcp, dstp, e0, c);
    // msg GVP 0 (384 -> 128, act)
    gemm(mvcat, 384, 3 * c, m0wh, 384, 384, 384, nullptr, 0, vh0, 384, nullptr, 0);
    k_vnorm<<<dim3(3, c), 128, 0, stream>>>(vh0, p384, 384, scat0, 768, 384);
    gemm(scat0, 768, c, m0ws, 768, 128, 768, P(CB + 2), 1, scat1, 256, nullptr, 0);
    gemm(vh0, 384, 3 * c, m0wv, 384, 128, 384, nullptr, 0, vt0, 128, nullptr, 0);
    k_vgate<<<((size_t)c * 128 + 255) / 256, 256, 0, stream>>>(vt0, p128, p128, v1, nullptr);
    // msg GVP 1 (128 -> 128, act)
    gemm(v1, 128, 3 * c, m1wh, 128, 128, 128, nullptr, 0, vh1, 128, nullptr, 0);
    k_vnorm<<<dim3(1, c), 128, 0, stream>>>(vh1, p128, 128, scat1, 256, 128);
    gemm(scat1, 256, c, m1ws, 256, 128, 256, P(CB + 6), 1, scat2, 256, nullptr, 0);
    gemm(vh1, 128, 3 * c, m1wv, 128, 128, 128, nullptr, 0, vt0, 128, nullptr, 0);
    k_vgate<<<((size_t)c * 128 + 255) / 256, 256, 0, stream>>>(vt0, p128, p128, v2, nullptr);
    // msg GVP 2 (128 -> 128, no act)
    gemm(v2, 128, 3 * c, m2wh, 128, 128, 128, nullptr, 0, vh2, 128, nullptr, 0);
    k_vnorm<<<dim3(1, c), 128, 0, stream>>>(vh2, p128, 128, scat2, 256, 128);
    gemm(scat2, 256, c, m2ws, 256, 128, 256, P(CB + 10), 0, nullptr, 0, msO, 128);
    gemm(vh2, 128, 3 * c, m2wv, 128, 128, 128, nullptr, 0, nullptr, 0, mvO, 128);
    k_scatter<<<(c * 128 + 255) / 256, 256, 0, stream>>>(
        msO, mvO, p128, dstp, e0, c, asum, avum, (size_t)NN * 128);
  }

  // ---- residual + ln0 ----
  k_ln0<<<(NN + 7) / 8, 256, 0, stream>>>(sN32, asum, vN32, avum, (size_t)NN * 128,
                                          cntb, P(CB + 12), P(CB + 13),
                                          sln32, scatF0, 384, vln32, vln16, NN);

  // ---- feed-forward GVPs (nodes) ----
  _Float16* vhf0  = (_Float16*)alloc(3ull * NN * 256 * 2);
  _Float16* scatF1 = (_Float16*)alloc((size_t)NN * 768 * 2);  // [ds0 | vn_ff1]
  _Float16* vof0  = (_Float16*)alloc(3ull * NN * 256 * 2);
  _Float16* dv0   = (_Float16*)alloc(3ull * NN * 256 * 2);
  _Float16* vhf1  = (_Float16*)alloc(3ull * NN * 256 * 2);

  gemm(vln16, 128, 3 * NN, f0wh, 128, 256, 128, nullptr, 0, vhf0, 256, nullptr, 0);
  k_vnorm<<<dim3(2, NN), 128, 0, stream>>>(vhf0, (size_t)NN * 256, 256, scatF0, 384, 128);
  gemm(scatF0, 384, NN, f0ws, 384, 512, 384, P(CB + 16), 1, scatF1, 768, nullptr, 0);
  gemm(vhf0, 256, 3 * NN, f0wv, 256, 256, 256, nullptr, 0, vof0, 256, nullptr, 0);
  k_vgate<<<((size_t)NN * 256 + 255) / 256, 256, 0, stream>>>(
      vof0, (size_t)NN * 256, (size_t)NN * 256, dv0, nullptr);
  gemm(dv0, 256, 3 * NN, f1wh, 256, 256, 256, nullptr, 0, vhf1, 256, nullptr, 0);
  k_vnorm<<<dim3(2, NN), 128, 0, stream>>>(vhf1, (size_t)NN * 256, 256, scatF1, 768, 512);
  gemm(scatF1, 768, NN, f1ws, 768, 128, 768, P(CB + 20), 0, nullptr, 0, ds1, 128);
  // ff1 vector output + ln1 vector path skipped (dead).

  k_ln1<<<(NN + 7) / 8, 256, 0, stream>>>(sln32, ds1, P(CB + 22), P(CB + 23), xs, NN);
  k_pool<<<(NN * 128 + 255) / 256, 256, 0, stream>>>(xs, batch, gsum, gcnt, NN);
  k_mlp<<<1, 64, 0, stream>>>(gsum, gcnt, P(60), P(61), P(62), P(63), P(64), P(65),
                              (float*)d_out);
}
